// SelfAttentionMemoryBank_40166534152224
// MI455X (gfx1250) — matrix operand order repaired and checked by reading, not compiled, not run on hardware
//
#include <hip/hip_runtime.h>

// ---------------------------------------------------------------------------
// SelfAttentionMemoryBank for MI455X (gfx1250, wave32, WMMA f16->f32).
//
//   xm   = [x ; mem^T]                         (B,S,D), S=2304
//   qkv  = xm @ W^T    (f16 WMMA, f32 accum)
//   C_j  = log2-domain column softmax stats of logits = k q^T / sqrt(D)
//   out  = sum_j exp2(l[i,j]*K - C_j) * v[j,:]   (E recomputed, never stored)
//
// k tiles are staged into LDS with GLOBAL_LOAD_ASYNC_TO_LDS_B128 (ASYNCcnt),
// double-buffered in the stats pass so the L2->LDS DMA overlaps the WMMAs.
// ---------------------------------------------------------------------------

typedef _Float16 h16;
typedef __attribute__((ext_vector_type(8)))  _Float16 v8h;
typedef __attribute__((ext_vector_type(16))) _Float16 v16h;
typedef __attribute__((ext_vector_type(8)))  float    v8f;

static constexpr int B_  = 8;
static constexpr int N_  = 2048;
static constexpr int M_  = 256;
static constexpr int S_  = 2304;   // N_ + M_
static constexpr int D_  = 512;
static constexpr int O3_ = 1536;   // 3*D_

// LDS row strides (in h16 elements): keep ds_load_b128 16B-aligned and
// spread rows across banks (stride/2 dwords % 64 == 4).
static constexpr int KLD = 520;    // for staged k tiles (512 + 8)
static constexpr int ELD = 136;    // for staged E tiles (128 + 8)

// log2(e)/sqrt(D): move logits into base-2 exponent domain
#define KSCALE (1.4426950408889634f / 22.62741699796952f)

#define DEVINL __device__ __forceinline__

DEVINL v8f wmma_f16(v16h a, v16h b, v8f c) {
  // D = A(16x32,f16) * B(32x16,f16) + C(16x16,f32)
  return __builtin_amdgcn_wmma_f32_16x16x32_f16(false, a, false, b,
                                                (short)0, c, false, false);
}

// A fragment 16x32 (MxK) from row-major f16 src (leading dim ld, elems).
// ISA layout: lanes 0-15 hold row m, K={0..7,16..23}; lanes 16-31 K={8..15,24..31}.
DEVINL v16h load_frag_a(const h16* src, int ld, int m0, int k0) {
  const int lane = threadIdx.x & 31;
  const int hi   = lane >> 4;
  const h16* p = src + (size_t)(m0 + (lane & 15)) * ld + k0 + hi * 8;
  v8h lo = *reinterpret_cast<const v8h*>(p);
  v8h hp = *reinterpret_cast<const v8h*>(p + 16);
  v16h a;
#pragma unroll
  for (int i = 0; i < 8; ++i) { a[i] = lo[i]; a[8 + i] = hp[i]; }
  return a;
}

// B fragment 32x16 (KxN) where B[k][n] = src[n0+n][k0+k] (src rows = N index).
// ISA layout: lanes 0-15 hold col n, K={0..7,16..23}; lanes 16-31 K={8..15,24..31}.
DEVINL v16h load_frag_b(const h16* src, int ld, int n0, int k0) {
  const int lane = threadIdx.x & 31;
  const int hi   = lane >> 4;
  const h16* p = src + (size_t)(n0 + (lane & 15)) * ld + k0 + hi * 8;
  v8h lo = *reinterpret_cast<const v8h*>(p);
  v8h hp = *reinterpret_cast<const v8h*>(p + 16);
  v16h b;
#pragma unroll
  for (int i = 0; i < 8; ++i) { b[i] = lo[i]; b[8 + i] = hp[i]; }
  return b;
}

// Cooperative async copy of `rows` x 512 h16 (row-major, ld=512) into LDS
// (padded row stride KLD) using GLOBAL_LOAD_ASYNC_TO_LDS_B128.
// ISA: flat addr[31:0] inside the shared aperture IS the LDS byte address,
// so the low 32 bits of the generic __shared__ pointer feed VDST directly.
// Tracked by ASYNCcnt; caller must s_wait_asynccnt + barrier before reading.
DEVINL void async_stage_rows(h16* dst, const h16* src, int rows) {
  for (int idx = threadIdx.x; idx < rows * 64; idx += 256) {
    const int r = idx >> 6, c = (idx & 63) * 8;
    const uint32_t la = (uint32_t)(uintptr_t)(dst + r * KLD + c);
    const uint64_t ga = (uint64_t)(uintptr_t)(src + (size_t)r * D_ + c);
    asm volatile("global_load_async_to_lds_b128 %0, %1, off"
                 :: "v"(la), "v"(ga) : "memory");
  }
}

DEVINL void wait_async0() {
  asm volatile("s_wait_asynccnt 0x0" ::: "memory");
}

// ---------------------------------------------------------------------------
// K0: pack xm = [x ; mem^T] to f16, and W to f16
// ---------------------------------------------------------------------------
__global__ void k_pack_xm(const float* __restrict__ x, const float* __restrict__ mem,
                          h16* __restrict__ xm) {
  size_t idx = (size_t)blockIdx.x * blockDim.x + threadIdx.x;
  if (idx >= (size_t)B_ * S_ * D_) return;
  const int d  = (int)(idx & (D_ - 1));
  const size_t sd = idx >> 9;
  const int s = (int)(sd % S_);
  const int b = (int)(sd / S_);
  const float v = (s < N_) ? x[((size_t)b * N_ + s) * D_ + d]
                           : mem[(size_t)d * M_ + (s - N_)];   // mem (1,D,M)
  xm[idx] = (h16)v;
}

__global__ void k_pack_w(const float* __restrict__ w, h16* __restrict__ wh) {
  const int idx = blockIdx.x * blockDim.x + threadIdx.x;
  if (idx < O3_ * D_) wh[idx] = (h16)w[idx];
}

// ---------------------------------------------------------------------------
// K1: qkv GEMM. One wave computes a 32(s) x 64(o) block = 2x4 WMMA tiles.
// q,k stored row-major (s,d) f16; v stored transposed (d,s) f16 so it can be
// a contiguous B-operand in pass 2.
// ---------------------------------------------------------------------------
__global__ void __launch_bounds__(256) k_qkv(const h16* __restrict__ xm,
                                             const h16* __restrict__ wh,
                                             h16* __restrict__ qh,
                                             h16* __restrict__ kh,
                                             h16* __restrict__ vT) {
  const int wave = threadIdx.x >> 5;
  const int lane = threadIdx.x & 31;
  int task = blockIdx.x * 8 + wave;            // 8 * 72 * 24 wave tasks
  const int ob = task % 24; task /= 24;
  const int sb = task % 72; const int b = task / 72;
  const h16* xb = xm + (size_t)b * S_ * D_;
  const int s0 = sb * 32, o0 = ob * 64;

  v8f acc[2][4] = {};
  for (int k0 = 0; k0 < D_; k0 += 32) {
    v16h a0 = load_frag_a(xb, D_, s0,      k0);
    v16h a1 = load_frag_a(xb, D_, s0 + 16, k0);
#pragma unroll
    for (int j = 0; j < 4; ++j) {
      v16h bf = load_frag_b(wh, D_, o0 + 16 * j, k0);  // B[k][o] = W[o][k]
      acc[0][j] = wmma_f16(a0, bf, acc[0][j]);
      acc[1][j] = wmma_f16(a1, bf, acc[1][j]);
    }
  }

  const int n = lane & 15, hi = lane >> 4;
#pragma unroll
  for (int t = 0; t < 2; ++t) {
#pragma unroll
    for (int j = 0; j < 4; ++j) {
      const int o = o0 + 16 * j + n;
      const int srow = s0 + 16 * t + hi * 8;
      if (o < D_) {                 // Q, row-major (s,d)
        h16* dst = qh + ((size_t)b * S_ + srow) * D_ + o;
#pragma unroll
        for (int r = 0; r < 8; ++r) dst[(size_t)r * D_] = (h16)acc[t][j][r];
      } else if (o < 2 * D_) {      // K, row-major (s,d)
        h16* dst = kh + ((size_t)b * S_ + srow) * D_ + (o - D_);
#pragma unroll
        for (int r = 0; r < 8; ++r) dst[(size_t)r * D_] = (h16)acc[t][j][r];
      } else {                      // V transposed (d,s): contiguous in s
        v8h hv;
#pragma unroll
        for (int r = 0; r < 8; ++r) hv[r] = (h16)acc[t][j][r];
        *reinterpret_cast<v8h*>(vT + ((size_t)b * D_ + (o - 2 * D_)) * S_ + srow) = hv;
      }
    }
  }
}

// ---------------------------------------------------------------------------
// K2: column softmax stats. Block = 8 waves sharing (b, 128-wide j-block);
// wave w owns j-tile j0 = jb*128 + w*16 and keeps its 16 q B-fragments in
// VGPRs. The k i-tiles are double-buffered in LDS via async-to-LDS DMA so
// the copy of tile t+1 overlaps the 16 WMMAs of tile t. Online max/sum in
// the f32 D-fragment; lane L owns column j0+(L&15) with rows split across
// L / L^16; merged via shfl_xor. Emits C_j = max2 + log2(sum 2^(l2-max2)).
// ---------------------------------------------------------------------------
__global__ void __launch_bounds__(256) k_colstats(const h16* __restrict__ qh,
                                                  const h16* __restrict__ kh,
                                                  float* __restrict__ Cst) {
  __shared__ h16 ksm[2][16 * KLD];             // 2 x 16.6 KB
  const int wave = threadIdx.x >> 5;
  const int lane = threadIdx.x & 31;
  const int jb = blockIdx.x % 18;              // grid = 8 * 18
  const int b  = blockIdx.x / 18;
  const int j0 = jb * 128 + wave * 16;
  const h16* qb = qh + (size_t)b * S_ * D_;
  const h16* kb = kh + (size_t)b * S_ * D_;

  v16h bq[16];
#pragma unroll
  for (int c = 0; c < 16; ++c) bq[c] = load_frag_b(qb, D_, j0, c * 32);

  async_stage_rows(ksm[0], kb, 16);            // prime the pipeline

  float m = -1e30f, z = 0.0f;
  for (int it = 0; it < S_ / 16; ++it) {
    const h16* cur = ksm[it & 1];
    wait_async0();                             // tile `it` landed in LDS
    __syncthreads();                           // ... for every wave
    if (it + 1 < S_ / 16)                      // DMA next tile during compute
      async_stage_rows((h16*)ksm[(it + 1) & 1], kb + (size_t)(it + 1) * 16 * D_, 16);
    v8f l = {};
#pragma unroll
    for (int c = 0; c < 16; ++c)
      l = wmma_f16(load_frag_a(cur, KLD, 0, c * 32), bq[c], l);
#pragma unroll
    for (int r = 0; r < 8; ++r) {
      const float t = l[r] * KSCALE;
      if (t > m) { z = z * __builtin_exp2f(m - t) + 1.0f; m = t; }
      else       { z += __builtin_exp2f(t - m); }
    }
  }
  const float mo = __shfl_xor(m, 16, 32);
  const float zo = __shfl_xor(z, 16, 32);
  const float mm = fmaxf(m, mo);
  const float zz = z * __builtin_exp2f(m - mm) + zo * __builtin_exp2f(mo - mm);
  if (lane < 16) Cst[b * S_ + j0 + lane] = mm + __builtin_log2f(zz);
}

// ---------------------------------------------------------------------------
// K3: out = E @ v with E recomputed in tiles. Block = (b, 32-row i-block);
// k block async-staged into LDS once. Per 128-wide j-step:
//   phase A: wave w computes the two 16x16 logit tiles of its own 16-col
//            slab (no redundant logits), applies exp2 with per-column C_j,
//            writes E into shared padded LDS (D-layout -> row-major).
//   phase B: wave w owns a 64-col d-slice: 8 f32 accumulator tiles,
//            A = E from LDS (shared), B = contiguous v^T fragments.
// ---------------------------------------------------------------------------
__global__ void __launch_bounds__(256) k_attnout(const h16* __restrict__ qh,
                                                 const h16* __restrict__ kh,
                                                 const h16* __restrict__ vT,
                                                 const float* __restrict__ Cst,
                                                 float* __restrict__ out) {
  __shared__ h16 ksm[32 * KLD];                // 33.3 KB
  __shared__ h16 esm[32 * ELD];                //  8.7 KB
  const int wave = threadIdx.x >> 5;
  const int lane = threadIdx.x & 31;
  const int ib = blockIdx.x % 36;              // grid = 8 * 36
  const int b  = blockIdx.x / 36;
  const int i0 = ib * 32;
  const h16* qb = qh + (size_t)b * S_ * D_;
  const h16* kb = kh + (size_t)b * S_ * D_;
  const h16* vb = vT + (size_t)b * D_ * S_;
  const float* Cb = Cst + b * S_;
  const int n = lane & 15, hi = lane >> 4;

  async_stage_rows(ksm, kb + (size_t)i0 * D_, 32);
  wait_async0();
  __syncthreads();

  v8f acc[8] = {};                             // [it 0..1][dt 0..3]
  for (int j0 = 0; j0 < S_; j0 += 128) {
    // ---- phase A: logits + exp2 for this wave's 16-column slab
    const int jA = j0 + wave * 16;
    v8f l0 = {}, l1 = {};
#pragma unroll
    for (int c = 0; c < 16; ++c) {
      v16h bf = load_frag_b(qb, D_, jA, c * 32);
      l0 = wmma_f16(load_frag_a(ksm, KLD, 0,  c * 32), bf, l0);
      l1 = wmma_f16(load_frag_a(ksm, KLD, 16, c * 32), bf, l1);
    }
    const float cj = Cb[jA + n];
#pragma unroll
    for (int r = 0; r < 8; ++r) {
      esm[(hi * 8 + r) * ELD + wave * 16 + n] =
          (h16)__builtin_exp2f(l0[r] * KSCALE - cj);
      esm[(16 + hi * 8 + r) * ELD + wave * 16 + n] =
          (h16)__builtin_exp2f(l1[r] * KSCALE - cj);
    }
    __syncthreads();
    // ---- phase B: accumulate this wave's 64-col d-slice
    const int d0w = wave * 64;
#pragma unroll
    for (int c = 0; c < 4; ++c) {              // K chunks of 32 inside j-step
      v16h ae0 = load_frag_a(esm, ELD, 0,  c * 32);
      v16h ae1 = load_frag_a(esm, ELD, 16, c * 32);
#pragma unroll
      for (int dt = 0; dt < 4; ++dt) {         // B[k=j][n=d] = vT[d][j]
        v16h bv = load_frag_b(vb, S_, d0w + dt * 16, j0 + c * 32);
        acc[dt]     = wmma_f16(ae0, bv, acc[dt]);
        acc[4 + dt] = wmma_f16(ae1, bv, acc[4 + dt]);
      }
    }
    __syncthreads();                           // protect esm before overwrite
  }

#pragma unroll
  for (int t = 0; t < 2; ++t) {
#pragma unroll
    for (int dt = 0; dt < 4; ++dt) {
      float* dst = out + ((size_t)b * S_ + i0 + t * 16 + hi * 8) * D_
                       + wave * 64 + dt * 16 + n;
#pragma unroll
      for (int r = 0; r < 8; ++r) dst[(size_t)r * D_] = acc[t * 4 + dt][r];
    }
  }
}

// ---------------------------------------------------------------------------
extern "C" void kernel_launch(void* const* d_in, const int* in_sizes, int n_in,
                              void* d_out, int out_size, void* d_ws, size_t ws_size,
                              hipStream_t stream) {
  const float* x   = (const float*)d_in[0];   // (8,2048,512)
  const float* mem = (const float*)d_in[1];   // (1,512,256)
  const float* w   = (const float*)d_in[2];   // (1536,512)
  float* out = (float*)d_out;                 // (8,2304,512) f32
  (void)in_sizes; (void)n_in; (void)out_size; (void)ws_size;

  char* ws = (char*)d_ws;
  size_t off = 0;
  auto take = [&](size_t bytes) {
    char* p = ws + off;
    off += (bytes + 255) & ~(size_t)255;
    return p;
  };
  h16*   wh  = (h16*)take((size_t)O3_ * D_ * sizeof(h16));     //  1.5 MB
  h16*   xm  = (h16*)take((size_t)B_ * S_ * D_ * sizeof(h16)); // 18.9 MB
  h16*   qh  = (h16*)take((size_t)B_ * S_ * D_ * sizeof(h16)); // 18.9 MB
  h16*   kh  = (h16*)take((size_t)B_ * S_ * D_ * sizeof(h16)); // 18.9 MB
  h16*   vT  = (h16*)take((size_t)B_ * D_ * S_ * sizeof(h16)); // 18.9 MB
  float* Cst = (float*)take((size_t)B_ * S_ * sizeof(float));  // 74 KB

  k_pack_xm<<<(B_ * S_ * D_ + 255) / 256, 256, 0, stream>>>(x, mem, xm);
  k_pack_w<<<(O3_ * D_ + 255) / 256, 256, 0, stream>>>(w, wh);
  k_qkv<<<(8 * 72 * 24) / 8, 256, 0, stream>>>(xm, wh, qh, kh, vT);
  k_colstats<<<8 * 18, 256, 0, stream>>>(qh, kh, Cst);
  k_attnout<<<8 * 36, 256, 0, stream>>>(qh, kh, vT, Cst, out);
}
